// Net_88596585382862
// MI455X (gfx1250) — matrix operand order repaired and checked by reading, not compiled, not run on hardware
//
#include <hip/hip_runtime.h>

// Problem dims
#define NSTEPS 50
#define NIN    784
#define NINP   800     // K padded to 25*32 for layer-1 GEMM
#define NHID   1000
#define NHIDP  1024    // N padded (layer1) / K padded (layer2)
#define NOUT   10
#define NOUTP  16
#define BATCH  1024
#define KT1    25      // k-tiles (32 wide) for GEMM1
#define NT1    64      // n-tiles (16 wide) for GEMM1 (padded N=1024)
#define KT2    32      // k-tiles for GEMM2 (padded K=1024)

#define B1_SLICE (KT1 * 32 * 16)   // 12800 bf16 elems per nt column-slice
#define B2_SLICE (KT2 * 32 * 16)   // 16384 bf16 elems (all of padded w2)

typedef __bf16  bf16_t;
typedef bf16_t  v16bf __attribute__((ext_vector_type(16)));
typedef float   v8f   __attribute__((ext_vector_type(8)));
typedef unsigned int v4u __attribute__((ext_vector_type(4)));
typedef int          v8i __attribute__((ext_vector_type(8)));
typedef int          v4i __attribute__((ext_vector_type(4)));

union Frag {
  v16bf v;
  uint4 q[2];
  unsigned short u[16];
};

__device__ __forceinline__ unsigned short f2bf_rne(float f) {
  unsigned int u = __float_as_uint(f);
  u += 0x7FFFu + ((u >> 16) & 1u);   // round-to-nearest-even (finite values)
  return (unsigned short)(u >> 16);
}

// Load 16 contiguous bf16 (32 bytes) — B fragment per lane (global or LDS)
__device__ __forceinline__ v16bf load_frag16(const unsigned short* p) {
  Frag f;
  f.q[0] = ((const uint4*)p)[0];
  f.q[1] = ((const uint4*)p)[1];
  return f.v;
}

// A fragment per lane: elems 0..15 contiguous at p
__device__ __forceinline__ v16bf load_fragA(const unsigned short* p) {
  Frag f;
  f.q[0] = *(const uint4*)p;
  f.q[1] = *(const uint4*)(p + 8);
  return f.v;
}

// ---------------------------------------------------------------------------
// Tensor Data Mover: 1-D contiguous copy global -> LDS.
// D# per CDNA5 ISA: count=1, type=2 ("image"),
// data_size=3 (8-byte units), tensor_dim0 = tile_dim0 = units, dims1 = 1.
// units8 = bytes/8 (must fit tile_dim0's 16 bits).
// ---------------------------------------------------------------------------
#if __has_builtin(__builtin_amdgcn_tensor_load_to_lds)
#define HAVE_TDM 1
#if __has_include(<hip/amd_detail/amd_gfx1250_TDM.h>)
#define TDM_6ARG 1   // amdgpu-toolchain (clang-23 + therock headers): 6-arg form
#endif

__device__ __forceinline__ void tdm_load_1d(void* lds_dst, const void* gsrc,
                                            unsigned units8) {
  // Generic->LDS address: shared-aperture flat addresses carry the LDS byte
  // offset in the low 32 bits (ISA 10.2 aperture mapping).
  unsigned lds_addr = (unsigned)(uintptr_t)lds_dst;
  unsigned long long ga = (unsigned long long)(uintptr_t)gsrc;

  v4u g0;
  g0.x = 1u;                                        // count=1 (valid user D#)
  g0.y = lds_addr;                                  // bits 63:32 lds_addr
  g0.z = (unsigned)(ga & 0xFFFFFFFFu);              // global_addr[31:0]
  g0.w = (unsigned)((ga >> 32) & 0x01FFFFFFu)       // global_addr[56:32]
       | (2u << 30);                                // type=2

  v8i g1;
  g1[0] = (int)(3u << 16);                          // data_size=3 (8B), mask=0
  g1[1] = (int)((units8 & 0xFFFFu) << 16);          // tensor_dim0[15:0] @63:48
  g1[2] = (int)(((units8 >> 16) & 0xFFFFu)          // tensor_dim0[31:16]
              | (1u << 16));                        // tensor_dim1 = 1
  g1[3] = (int)((units8 & 0xFFFFu) << 16);          // tile_dim0 @127:112
  g1[4] = 1;                                        // tile_dim1 = 1
  g1[5] = (int)units8;                              // tensor_dim0_stride lo32
  g1[6] = 0;                                        // stride0 hi / stride1 lo
  g1[7] = 0;                                        // tensor_dim1_stride hi

  v4i z4 = {0, 0, 0, 0};
#if defined(TDM_6ARG)
  v8i z8 = {0, 0, 0, 0, 0, 0, 0, 0};
  __builtin_amdgcn_tensor_load_to_lds(g0, g1, z4, z4, z8, 0);
#else
  __builtin_amdgcn_tensor_load_to_lds(g0, g1, z4, z4, 0);
#endif
}
#endif // HAVE_TDM

// Block-level stage of nElems bf16 from global to LDS. Caller must
// __syncthreads() after all stages. TDM issued by wave 0 only (TENSORcnt is
// per-wave); fallback is a cooperative vector copy.
__device__ __forceinline__ void stage_to_lds(unsigned short* dstLds,
                                             const unsigned short* src,
                                             int nElems) {
#if defined(HAVE_TDM)
  if (threadIdx.x < 32) {
    tdm_load_1d(dstLds, src, (unsigned)(nElems >> 2));  // bytes/8
    __builtin_amdgcn_s_wait_tensorcnt(0);
  }
#else
  const uint4* s = (const uint4*)src;
  uint4* d = (uint4*)dstLds;
  int n16 = nElems >> 3;                              // 16B chunks
  for (int i = threadIdx.x; i < n16; i += blockDim.x) d[i] = s[i];
#endif
}

// ---------------------------------------------------------------------------
// Weight swizzle: split fp32 -> (bf16 hi, bf16 lo). Layout [nt][kt][lane][e]
// so one nt column-slice (all K) is contiguous for a single TDM descriptor.
// lane: n = nt*16+(lane&15), k = kt*32 + (lane>=16 ? 16 : 0) + e.
// ---------------------------------------------------------------------------
__global__ __launch_bounds__(256) void prep_w1(const float* __restrict__ w1,
                                               unsigned short* __restrict__ hi,
                                               unsigned short* __restrict__ lo) {
  int idx = blockIdx.x * 256 + threadIdx.x;      // < NT1*KT1*32*16 = 819200
  int nt  = idx / B1_SLICE;
  int rem = idx - nt * B1_SLICE;
  int kt  = rem >> 9;
  int L   = (rem >> 4) & 31;
  int e   = rem & 15;
  int k = kt * 32 + ((L & 16) ? 16 : 0) + e;
  int n = nt * 16 + (L & 15);
  float w = (k < NIN && n < NHID) ? w1[n * NIN + k] : 0.f;
  unsigned short h = f2bf_rne(w);
  float hf = __uint_as_float(((unsigned int)h) << 16);
  unsigned short l = f2bf_rne(w - hf);
  hi[idx] = h;
  lo[idx] = l;
}

__global__ __launch_bounds__(256) void prep_w2(const float* __restrict__ w2,
                                               unsigned short* __restrict__ hi,
                                               unsigned short* __restrict__ lo) {
  int idx = blockIdx.x * 256 + threadIdx.x;      // < KT2*32*16 = 16384
  int e  = idx & 15;
  int L  = (idx >> 4) & 31;
  int kt = idx >> 9;
  int k = kt * 32 + ((L & 16) ? 16 : 0) + e;
  int n = L & 15;
  float w = (k < NHID && n < NOUT) ? w2[n * NHID + k] : 0.f;
  unsigned short h = f2bf_rne(w);
  float hf = __uint_as_float(((unsigned int)h) << 16);
  unsigned short l = f2bf_rne(w - hf);
  hi[idx] = h;
  lo[idx] = l;
}

// ---------------------------------------------------------------------------
// Layer 0 LIF (elementwise over [1024 x 784]); writes m0 and bf16 spikes into
// K-padded spike buffer. Prefetch next timestep's x slab (global_prefetch_b8).
// ---------------------------------------------------------------------------
__global__ __launch_bounds__(256) void lif0(const float* __restrict__ xt,
                                            const float* __restrict__ xnext,
                                            const float* __restrict__ thr0,
                                            const float* __restrict__ beta0p,
                                            float* __restrict__ m0,
                                            unsigned short* __restrict__ spk0) {
  int i = blockIdx.x * 256 + threadIdx.x;        // exactly 802816 threads
  int row = i / NIN;
  int k = i - row * NIN;
  __builtin_prefetch(xnext + i, 0, 1);
  float beta = beta0p[0];
  float xv = xt[i];
  float m  = m0[i];
  float tk = thr0[k];
  float reset = (m - tk) > 0.f ? 1.f : 0.f;
  float mn = (beta * m + xv) * (1.f - reset);
  m0[i] = mn;
  spk0[row * NINP + k] = (mn - tk) > 0.f ? (unsigned short)0x3F80u
                                         : (unsigned short)0u;
}

// ---------------------------------------------------------------------------
// GEMM1 (spk0[1024x784] x w1^T[784x1000]) via split-bf16 WMMA, fused
// bias + ReLU + LIF1 epilogue. Block = 8 waves sharing one N-tile (8 M-tiles);
// the block's B column-slice (hi+lo, 51.2KB) is TDM-staged to LDS once.
// ---------------------------------------------------------------------------
__global__ __launch_bounds__(256) void gemm1_lif1(const unsigned short* __restrict__ spk0,
                                                  const unsigned short* __restrict__ w1hi,
                                                  const unsigned short* __restrict__ w1lo,
                                                  const float* __restrict__ b1,
                                                  const float* __restrict__ thr1,
                                                  const float* __restrict__ beta1p,
                                                  float* __restrict__ m1,
                                                  unsigned short* __restrict__ spk1) {
  __shared__ __align__(16) unsigned short sBhi[B1_SLICE];
  __shared__ __align__(16) unsigned short sBlo[B1_SLICE];

  const int lane = threadIdx.x & 31;
  const int warp = threadIdx.x >> 5;
  const int nt   = blockIdx.x & 63;                 // shared by all 8 waves
  const int mt   = ((blockIdx.x >> 6) << 3) + warp; // 8 M-tiles per block
  const int half = lane >> 4;
  const int r    = lane & 15;

  stage_to_lds(sBhi, w1hi + (size_t)nt * B1_SLICE, B1_SLICE);
  stage_to_lds(sBlo, w1lo + (size_t)nt * B1_SLICE, B1_SLICE);
  __syncthreads();

  v8f c = {0.f, 0.f, 0.f, 0.f, 0.f, 0.f, 0.f, 0.f};

  const int rowA = mt * 16 + r;
  const unsigned short* ab = spk0 + (size_t)rowA * NINP + half * 16;
  const int fb = lane * 16;

  for (int kt = 0; kt < KT1; ++kt) {
    v16bf a  = load_fragA(ab + kt * 32);
    v16bf bh = load_frag16(sBhi + kt * 512 + fb);
    v16bf bl = load_frag16(sBlo + kt * 512 + fb);
    c = __builtin_amdgcn_wmma_f32_16x16x32_bf16(false, a, false, bl,
                                                (short)0, c, false, false);
    c = __builtin_amdgcn_wmma_f32_16x16x32_bf16(false, a, false, bh,
                                                (short)0, c, false, false);
  }

  const int j = nt * 16 + r;          // output neuron (N dim)
  if (j < NHID) {
    float beta = beta1p[0];
    float bj = b1[j];
    float tj = thr1[j];
#pragma unroll
    for (int v = 0; v < 8; ++v) {
      int row = mt * 16 + v + half * 8;           // C/D layout: M = v (+8 hi half)
      float cur = fmaxf(c[v] + bj, 0.f);
      size_t idx = (size_t)row * NHIDP + j;
      float m = m1[idx];
      float reset = (m - tj) > 0.f ? 1.f : 0.f;
      float mn = (beta * m + cur) * (1.f - reset);
      m1[idx] = mn;
      spk1[idx] = (mn - tj) > 0.f ? (unsigned short)0x3F80u : (unsigned short)0u;
    }
  }
}

// ---------------------------------------------------------------------------
// GEMM2 (spk1[1024x1000] x w2^T[1000x10], N padded to 16) + LIF2, writes
// spk2/m2 straight into d_out ([50,1024,10] spikes then [50,1024,10] mems).
// All waves share the single N-tile; whole padded w2 (64KB) TDM-staged to LDS.
// ---------------------------------------------------------------------------
__global__ __launch_bounds__(256) void gemm2_lif2(const unsigned short* __restrict__ spk1,
                                                  const unsigned short* __restrict__ w2hi,
                                                  const unsigned short* __restrict__ w2lo,
                                                  const float* __restrict__ b2,
                                                  const float* __restrict__ thr2,
                                                  const float* __restrict__ beta2p,
                                                  float* __restrict__ m2,
                                                  float* __restrict__ out,
                                                  int t) {
  __shared__ __align__(16) unsigned short sBhi[B2_SLICE];
  __shared__ __align__(16) unsigned short sBlo[B2_SLICE];

  const int lane = threadIdx.x & 31;
  const int warp = threadIdx.x >> 5;
  const int mt   = (blockIdx.x << 3) + warp;        // 64 waves total
  const int half = lane >> 4;
  const int r    = lane & 15;

  stage_to_lds(sBhi, w2hi, B2_SLICE);
  stage_to_lds(sBlo, w2lo, B2_SLICE);
  __syncthreads();

  v8f c = {0.f, 0.f, 0.f, 0.f, 0.f, 0.f, 0.f, 0.f};

  const int rowA = mt * 16 + r;
  const unsigned short* ab = spk1 + (size_t)rowA * NHIDP + half * 16;
  const int fb = lane * 16;

  for (int kt = 0; kt < KT2; ++kt) {
    v16bf a  = load_fragA(ab + kt * 32);
    v16bf bh = load_frag16(sBhi + kt * 512 + fb);
    v16bf bl = load_frag16(sBlo + kt * 512 + fb);
    c = __builtin_amdgcn_wmma_f32_16x16x32_bf16(false, a, false, bl,
                                                (short)0, c, false, false);
    c = __builtin_amdgcn_wmma_f32_16x16x32_bf16(false, a, false, bh,
                                                (short)0, c, false, false);
  }

  const int j = r;                    // output class
  if (j < NOUT) {
    float beta = beta2p[0];
    float bj = b2[j];
    float tj = thr2[j];
    float* spk_out = out + (size_t)t * BATCH * NOUT;
    float* mem_out = out + (size_t)NSTEPS * BATCH * NOUT + (size_t)t * BATCH * NOUT;
#pragma unroll
    for (int v = 0; v < 8; ++v) {
      int row = mt * 16 + v + half * 8;
      float cur = fmaxf(c[v] + bj, 0.f);
      size_t midx = (size_t)row * NOUTP + j;
      float m = m2[midx];
      float reset = (m - tj) > 0.f ? 1.f : 0.f;
      float mn = (beta * m + cur) * (1.f - reset);
      m2[midx] = mn;
      float spk = (mn - tj) > 0.f ? 1.f : 0.f;
      spk_out[(size_t)row * NOUT + j] = spk;
      mem_out[(size_t)row * NOUT + j] = mn;
    }
  }
}

// ---------------------------------------------------------------------------
extern "C" void kernel_launch(void* const* d_in, const int* in_sizes, int n_in,
                              void* d_out, int out_size, void* d_ws, size_t ws_size,
                              hipStream_t stream) {
  const float* x     = (const float*)d_in[0];
  const float* w1    = (const float*)d_in[1];
  const float* b1    = (const float*)d_in[2];
  const float* w2    = (const float*)d_in[3];
  const float* b2    = (const float*)d_in[4];
  const float* thr0  = (const float*)d_in[5];
  const float* thr1  = (const float*)d_in[6];
  const float* thr2  = (const float*)d_in[7];
  const float* beta0 = (const float*)d_in[8];
  const float* beta1 = (const float*)d_in[9];
  const float* beta2 = (const float*)d_in[10];

  char* ws = (char*)d_ws;
  size_t off = 0;
  // --- persistent per-launch state (zeroed every call) ---
  float* m0 = (float*)(ws + off); off += (size_t)BATCH * NIN   * sizeof(float);
  float* m1 = (float*)(ws + off); off += (size_t)BATCH * NHIDP * sizeof(float);
  float* m2 = (float*)(ws + off); off += (size_t)BATCH * NOUTP * sizeof(float);
  size_t state_bytes = off;
  // --- swizzled split-bf16 weights (rebuilt every call) ---
  unsigned short* w1hi = (unsigned short*)(ws + off); off += (size_t)NT1 * B1_SLICE * 2;
  unsigned short* w1lo = (unsigned short*)(ws + off); off += (size_t)NT1 * B1_SLICE * 2;
  unsigned short* w2hi = (unsigned short*)(ws + off); off += (size_t)B2_SLICE * 2;
  unsigned short* w2lo = (unsigned short*)(ws + off); off += (size_t)B2_SLICE * 2;
  // --- spike buffers (bf16, padded; padding must stay zero) ---
  size_t spk_off = off;
  unsigned short* spk0 = (unsigned short*)(ws + off); off += (size_t)BATCH * NINP  * 2;
  unsigned short* spk1 = (unsigned short*)(ws + off); off += (size_t)BATCH * NHIDP * 2;
  size_t spk_bytes = off - spk_off;

  // Zero membrane state + spike buffers (capture-safe stream ops).
  hipMemsetAsync(ws, 0, state_bytes, stream);
  hipMemsetAsync(ws + spk_off, 0, spk_bytes, stream);

  // Build split-bf16 swizzled weights.
  prep_w1<<<(NT1 * B1_SLICE) / 256, 256, 0, stream>>>(w1, w1hi, w1lo);
  prep_w2<<<B2_SLICE / 256, 256, 0, stream>>>(w2, w2hi, w2lo);

  for (int t = 0; t < NSTEPS; ++t) {
    const float* xt = x + (size_t)t * BATCH * NIN;
    const float* xn = x + (size_t)((t + 1 < NSTEPS) ? (t + 1) : t) * BATCH * NIN;
    lif0<<<(BATCH * NIN) / 256, 256, 0, stream>>>(xt, xn, thr0, beta0, m0, spk0);
    gemm1_lif1<<<(64 * NT1) / 8, 256, 0, stream>>>(spk0, w1hi, w1lo, b1, thr1,
                                                   beta1, m1, spk1);
    gemm2_lif2<<<64 / 8, 256, 0, stream>>>(spk1, w2hi, w2lo, b2, thr2, beta2, m2,
                                           (float*)d_out, t);
  }
}
